// SliceAttention_78554951844090
// MI455X (gfx1250) — matrix operand order repaired and checked by reading, not compiled, not run on hardware
//
#include <hip/hip_runtime.h>
#include <hip/hip_bf16.h>

typedef __bf16 bf16_t;
typedef __attribute__((ext_vector_type(16))) __bf16 v16bf;
typedef __attribute__((ext_vector_type(8)))  __bf16 v8bf;
typedef __attribute__((ext_vector_type(4)))  __bf16 v4bf;
typedef __attribute__((ext_vector_type(8)))  float  v8f;
typedef __attribute__((ext_vector_type(4)))  float  v4f;

#define B_DIM 8
#define S_LEN 2048
#define N_DIM 1024
#define ROWS  (B_DIM * S_LEN)   // 16384

// padded LDS row strides (elements); keep 16B alignment, skew banks
#define PAD_NROW (N_DIM + 8)    // 1032 elems = 2064 B rows (Q/K panels)
#define PAD_KROW (32 + 8)       // 40 elems = 80 B rows (GEMM k-chunk tiles)

__device__ __forceinline__ float fast_rcp(float x) {
    return __builtin_amdgcn_rcpf(x);
}

// ---------------------------------------------------------------------------
// Kernel 1: one pass over `outer` (64 MB): emit bf16 copy Xb AND row sums
// ---------------------------------------------------------------------------
__global__ void __launch_bounds__(256)
convert_x_kernel(const float* __restrict__ X, bf16_t* __restrict__ Xb,
                 float* __restrict__ r)
{
    const int wave = (blockIdx.x * blockDim.x + threadIdx.x) >> 5;
    const int lane = threadIdx.x & 31;
    const float* row = X  + (size_t)wave * N_DIM;
    bf16_t*     orow = Xb + (size_t)wave * N_DIM;

    float s = 0.f;
    #pragma unroll
    for (int it = 0; it < N_DIM / (32 * 4); ++it) {
        const int idx = (it * 32 + lane) * 4;
        v4f v = *(const v4f*)(row + idx);
        s += v.x + v.y + v.z + v.w;
        v4bf o;
        o.x = (__bf16)v.x; o.y = (__bf16)v.y;
        o.z = (__bf16)v.z; o.w = (__bf16)v.w;
        *(v4bf*)(orow + idx) = o;
    }
    #pragma unroll
    for (int off = 16; off > 0; off >>= 1) s += __shfl_xor(s, off, 32);
    if (lane == 0) r[wave] = s;
}

// ---------------------------------------------------------------------------
// Kernel 2: convert both weight matrices to bf16 (one shot)
// ---------------------------------------------------------------------------
__global__ void __launch_bounds__(256)
convert_w_kernel(const float* __restrict__ Wq, const float* __restrict__ Wk,
                 bf16_t* __restrict__ Wqb, bf16_t* __restrict__ Wkb)
{
    const size_t i = ((size_t)blockIdx.x * blockDim.x + threadIdx.x) * 4;
    v4f q = *(const v4f*)(Wq + i);
    v4f k = *(const v4f*)(Wk + i);
    v4bf oq, ok;
    oq.x = (__bf16)q.x; oq.y = (__bf16)q.y; oq.z = (__bf16)q.z; oq.w = (__bf16)q.w;
    ok.x = (__bf16)k.x; ok.y = (__bf16)k.y; ok.z = (__bf16)k.z; ok.w = (__bf16)k.w;
    *(v4bf*)(Wqb + i) = oq;
    *(v4bf*)(Wkb + i) = ok;
}

// ---------------------------------------------------------------------------
// Kernel 3: LDS-blocked fused projections, branchless pipelined staging.
// Block = 256 threads -> 128(M) x 64(N) output block for BOTH Q and K.
// Per 32-wide K step: A(128x32), Bq(64x32), Bk(64x32) live in LDS; each wave
// (4x2 grid) runs 8 WMMAs. Next chunk is prefetched into registers while the
// current chunk is being consumed (overlaps L2 latency with WMMA + barrier).
// ---------------------------------------------------------------------------
__global__ void __launch_bounds__(256)
proj_kernel(const bf16_t* __restrict__ Xb,
            const bf16_t* __restrict__ Wqb, const float* __restrict__ bq,
            const bf16_t* __restrict__ Wkb, const float* __restrict__ bk,
            bf16_t* __restrict__ Qb, bf16_t* __restrict__ Kb)
{
    __shared__ bf16_t At [128 * PAD_KROW];
    __shared__ bf16_t BqS[ 64 * PAD_KROW];
    __shared__ bf16_t BkS[ 64 * PAD_KROW];

    const int tid  = threadIdx.x;
    const int lane = tid & 31;
    const int wv   = tid >> 5;
    const int g    = lane >> 4;
    const int lm   = lane & 15;
    const int mb   = blockIdx.x >> 4;   // 0..127  (16384/128 row blocks)
    const int nb   = blockIdx.x & 15;   // 0..15   (1024/64 col blocks)
    const int wm   = wv >> 1;           // 0..3  -> 32 rows each
    const int wn   = wv & 1;            // 0..1  -> 32 cols each

    // ---- loop-invariant staging pointers (selected once, no branch in loop)
    const bf16_t* src;
    bf16_t*       dst;
    if (tid < 128) {                    // A panel: one row per thread
        src = Xb + (size_t)(mb * 128 + tid) * N_DIM;
        dst = At + tid * PAD_KROW;
    } else if (tid < 192) {             // Bq panel
        src = Wqb + (size_t)(nb * 64 + (tid - 128)) * N_DIM;
        dst = BqS + (tid - 128) * PAD_KROW;
    } else {                            // Bk panel
        src = Wkb + (size_t)(nb * 64 + (tid - 192)) * N_DIM;
        dst = BkS + (tid - 192) * PAD_KROW;
    }

    v8f accq[2][2] = {{{}, {}}, {{}, {}}};
    v8f acck[2][2] = {{{}, {}}, {{}, {}}};

    // prefetch chunk 0
    v16bf t0 = *(const v16bf*)(src);
    v16bf t1 = *(const v16bf*)(src + 16);

    for (int k0 = 0; k0 < N_DIM; k0 += 32) {
        __syncthreads();                 // previous chunk's readers done
        *(v16bf*)(dst)      = t0;
        *(v16bf*)(dst + 16) = t1;
        if (k0 + 32 < N_DIM) {           // prefetch next chunk (uniform branch)
            t0 = *(const v16bf*)(src + k0 + 32);
            t1 = *(const v16bf*)(src + k0 + 48);
        }
        __syncthreads();                 // chunk visible to all waves

        // ---- fragments from LDS ----
        v16bf a[2], fq[2], fk[2];
        #pragma unroll
        for (int mi = 0; mi < 2; ++mi) {
            const bf16_t* ar = At + (wm * 32 + mi * 16 + lm) * PAD_KROW;
            v8bf lo = *(const v8bf*)(ar + g * 16);
            v8bf hi = *(const v8bf*)(ar + g * 16 + 8);
            #pragma unroll
            for (int i = 0; i < 8; ++i) { a[mi][i] = lo[i]; a[mi][i + 8] = hi[i]; }
        }
        #pragma unroll
        for (int ni = 0; ni < 2; ++ni) {
            const int row = wn * 32 + ni * 16 + lm;
            fq[ni] = *(const v16bf*)(BqS + row * PAD_KROW + g * 16);
            fk[ni] = *(const v16bf*)(BkS + row * PAD_KROW + g * 16);
        }

        #pragma unroll
        for (int mi = 0; mi < 2; ++mi) {
            #pragma unroll
            for (int ni = 0; ni < 2; ++ni) {
                accq[mi][ni] = __builtin_amdgcn_wmma_f32_16x16x32_bf16(
                    false, a[mi], false, fq[ni], (short)0, accq[mi][ni], false, false);
                acck[mi][ni] = __builtin_amdgcn_wmma_f32_16x16x32_bf16(
                    false, a[mi], false, fk[ni], (short)0, acck[mi][ni], false, false);
            }
        }
    }

    // ---- epilogue: bias + sigmoid (v_exp + v_rcp) + bf16 store ----
    #pragma unroll
    for (int ni = 0; ni < 2; ++ni) {
        const int nn = nb * 64 + wn * 32 + ni * 16 + lm;
        const float biasq = bq[nn];
        const float biask = bk[nn];
        #pragma unroll
        for (int mi = 0; mi < 2; ++mi) {
            const size_t row0 = (size_t)(mb * 128 + wm * 32 + mi * 16 + 8 * g);
            #pragma unroll
            for (int v = 0; v < 8; ++v) {
                float xq = fast_rcp(1.0f + __expf(-(accq[mi][ni][v] + biasq)));
                float xk = fast_rcp(1.0f + __expf(-(acck[mi][ni][v] + biask)));
                const size_t off = (row0 + v) * N_DIM + nn;
                Qb[off] = (__bf16)xq;
                Kb[off] = (__bf16)xk;
            }
        }
    }
}

// ---------------------------------------------------------------------------
// Kernel 4: LDS-blocked flash attention + weighted rowsum.
// Block = 8 waves owns a 128-row q-block; full Q panel staged once in LDS
// (256 KB), each 16-row K tile staged once per block (32 KB, 290 KB total).
// K tile kt+1 is prefetched into registers while tile kt is consumed.
// Each wave owns one 16-row q-tile -> no cross-wave combine.
// ---------------------------------------------------------------------------
__global__ void __launch_bounds__(256)
attn_kernel(const bf16_t* __restrict__ Qb, const bf16_t* __restrict__ Kb,
            const float* __restrict__ r, const float* __restrict__ mask,
            float* __restrict__ out)
{
    extern __shared__ bf16_t smem[];
    bf16_t* Qs = smem;                       // 128 * PAD_NROW
    bf16_t* Ks = smem + 128 * PAD_NROW;      // 16 * PAD_NROW

    const int tid  = threadIdx.x;
    const int lane = tid & 31;
    const int wv   = tid >> 5;
    const int g    = lane >> 4;
    const int lm   = lane & 15;
    const int b    = blockIdx.x >> 4;        // batch
    const int qb   = blockIdx.x & 15;        // 128-row q block
    const size_t qrow0 = (size_t)b * S_LEN + qb * 128;

    // ---- stage the whole Q block once: 128 x 1024 bf16 ----
    for (int c = tid; c < 128 * (N_DIM / 8); c += 256) {   // 64 iters/thread
        const int row = c >> 7;
        const int ch  = (c & 127) * 8;
        *(v8bf*)(Qs + row * PAD_NROW + ch) =
            *(const v8bf*)(Qb + (qrow0 + row) * N_DIM + ch);
    }

    float m8[8], l8[8], a8[8];
    #pragma unroll
    for (int v = 0; v < 8; ++v) { m8[v] = -1e30f; l8[v] = 0.f; a8[v] = 0.f; }

    const bf16_t* aq = Qs + (wv * 16 + lm) * PAD_NROW;   // this wave's q rows

    // ---- loop-invariant K staging addresses:
    // thread covers rows (tid>>7)+2i, fixed 16B column chunk (tid&127)*8
    const int     r0   = tid >> 7;               // 0 or 1
    const int     ch   = (tid & 127) * 8;
    const bf16_t* ksrc = Kb + ((size_t)b * S_LEN + r0) * N_DIM + ch;
    bf16_t*       kdst = Ks + r0 * PAD_NROW + ch;

    // prefetch K tile 0 into registers (8 x 16B per thread)
    v8bf kreg[8];
    #pragma unroll
    for (int i = 0; i < 8; ++i)
        kreg[i] = *(const v8bf*)(ksrc + (size_t)(2 * i) * N_DIM);

    __syncthreads();   // Q panel ready

    for (int kt = 0; kt < 128; ++kt) {
        // commit prefetched tile to LDS, then prefetch the next one
        #pragma unroll
        for (int i = 0; i < 8; ++i)
            *(v8bf*)(kdst + 2 * i * PAD_NROW) = kreg[i];
        if (kt + 1 < 128) {
            const bf16_t* p = ksrc + (size_t)(kt + 1) * 16 * N_DIM;
            #pragma unroll
            for (int i = 0; i < 8; ++i)
                kreg[i] = *(const v8bf*)(p + (size_t)(2 * i) * N_DIM);
        }
        __syncthreads();                 // K tile visible

        // ---- 16x16 score tile: 32 WMMAs over n ----
        v8f acc = {};
        #pragma unroll 4
        for (int n0 = 0; n0 < N_DIM; n0 += 32) {
            v16bf a, bb;
            v8bf lo = *(const v8bf*)(aq + n0 + g * 16);
            v8bf hi = *(const v8bf*)(aq + n0 + g * 16 + 8);
            #pragma unroll
            for (int i = 0; i < 8; ++i) { a[i] = lo[i]; a[i + 8] = hi[i]; }
            bb = *(const v16bf*)(Ks + lm * PAD_NROW + n0 + g * 16);
            acc = __builtin_amdgcn_wmma_f32_16x16x32_bf16(
                      false, a, false, bb, (short)0, acc, false, false);
        }

        // ---- online softmax fused with weighted rowsum ----
        const size_t kk = (size_t)b * S_LEN + kt * 16;
        const float mk = mask[(size_t)b * S_LEN + kt * 16 + lm];
        const float rk = r[kk + lm];
        #pragma unroll
        for (int v = 0; v < 8; ++v) {
            float t  = acc[v] + mk;
            float rm = t;
            rm = fmaxf(rm, __shfl_xor(rm, 1, 32));
            rm = fmaxf(rm, __shfl_xor(rm, 2, 32));
            rm = fmaxf(rm, __shfl_xor(rm, 4, 32));
            rm = fmaxf(rm, __shfl_xor(rm, 8, 32));
            float mnew = fmaxf(m8[v], rm);
            float p  = __expf(t - mnew);
            float ps = p;
            float ws = p * rk;
            ps += __shfl_xor(ps, 1, 32);  ws += __shfl_xor(ws, 1, 32);
            ps += __shfl_xor(ps, 2, 32);  ws += __shfl_xor(ws, 2, 32);
            ps += __shfl_xor(ps, 4, 32);  ws += __shfl_xor(ws, 4, 32);
            ps += __shfl_xor(ps, 8, 32);  ws += __shfl_xor(ws, 8, 32);
            float sc = __expf(m8[v] - mnew);
            l8[v] = l8[v] * sc + ps;
            a8[v] = a8[v] * sc + ws;
            m8[v] = mnew;
        }
        __syncthreads();   // readers done before Ks is overwritten
    }

    // each wave owns its 16 rows; lanes in a half-group hold identical state
    if (lm == 0) {
        #pragma unroll
        for (int v = 0; v < 8; ++v) {
            out[qrow0 + wv * 16 + v + 8 * g] = a8[v] * fast_rcp(l8[v]);
        }
    }
}

// ---------------------------------------------------------------------------
extern "C" void kernel_launch(void* const* d_in, const int* in_sizes, int n_in,
                              void* d_out, int out_size, void* d_ws, size_t ws_size,
                              hipStream_t stream)
{
    const float* outer = (const float*)d_in[0];   // [8,2048,1024]
    const float* mask  = (const float*)d_in[1];   // [8,1,2048]
    const float* Wk    = (const float*)d_in[2];   // [1024,1024]
    const float* bk    = (const float*)d_in[3];   // [1024]
    const float* Wq    = (const float*)d_in[4];   // [1024,1024]
    const float* bq    = (const float*)d_in[5];   // [1024]
    float* out = (float*)d_out;                   // [8,2048]

    bf16_t* Qb  = (bf16_t*)d_ws;                          // 32 MB
    bf16_t* Kb  = Qb  + (size_t)ROWS * N_DIM;             // 32 MB
    bf16_t* Xb  = Kb  + (size_t)ROWS * N_DIM;             // 32 MB
    bf16_t* Wqb = Xb  + (size_t)ROWS * N_DIM;             // 2 MB
    bf16_t* Wkb = Wqb + (size_t)N_DIM * N_DIM;            // 2 MB
    float*  r   = (float*)(Wkb + (size_t)N_DIM * N_DIM);  // 64 KB

    convert_x_kernel<<<ROWS / 8, 256, 0, stream>>>(outer, Xb, r);
    convert_w_kernel<<<(N_DIM * N_DIM) / (256 * 4), 256, 0, stream>>>(Wq, Wk, Wqb, Wkb);
    // 128 row-blocks x 16 col-blocks
    proj_kernel<<<128 * 16, 256, 0, stream>>>(Xb, Wqb, bq, Wkb, bk, Qb, Kb);
    // 8 batches x 16 q-blocks, 290 KB dynamic LDS
    const size_t attn_lds = (size_t)(128 + 16) * PAD_NROW * sizeof(bf16_t);
    attn_kernel<<<B_DIM * 16, 256, attn_lds, stream>>>(Qb, Kb, r, mask, out);
}
